// Attention_80685255623337
// MI455X (gfx1250) — compile-verified, matrix operand order checked by reading, not run on hardware
//
#include <hip/hip_runtime.h>
#include <hip/hip_bf16.h>
#include <math.h>

typedef __bf16 bf16_t;
typedef __attribute__((ext_vector_type(16))) __bf16 v16bf;
typedef __attribute__((ext_vector_type(8)))  __bf16 v8bf;
typedef __attribute__((ext_vector_type(8)))  float  v8f;
typedef __attribute__((ext_vector_type(4)))  unsigned int u32x4;
typedef __attribute__((ext_vector_type(8)))  int i32x8;
typedef __attribute__((ext_vector_type(4)))  int i32x4;

constexpr int T_   = 4096;
constexpr int D_   = 2048;
constexpr int NH_  = 16;
constexpr int NKV_ = 4;
constexpr int DH_  = 128;
constexpr int KVD_ = NKV_ * DH_;   // 512
constexpr int GRP_ = NH_ / NKV_;   // 4

#ifndef __has_builtin
#define __has_builtin(x) 0
#endif
#if __has_builtin(__builtin_amdgcn_tensor_load_to_lds) && \
    __has_builtin(__builtin_amdgcn_s_wait_tensorcnt)
#define HAVE_TDM 1
#else
#define HAVE_TDM 0
#endif

// WMMA 16-bit operand fragment: VGPRs 0-3 = K(lanehalf*8 .. +8),
// VGPRs 4-7 = K(16+lanehalf*8 .. +8) per CDNA5 ISA.
union Frag16 {
  v16bf v;
  v8bf  h[2];
};

#if HAVE_TDM
// ---------------------------------------------------------------------------
// Issue a 2-D TDM tile load: tile_d1 rows x tile_d0 elements (2B each) from a
// row-major tensor with row stride stride0 (elements) into contiguous LDS.
// D# packing per CDNA5 ISA.
// ---------------------------------------------------------------------------
__device__ __forceinline__ void tdm_load_2d_bf16(
    unsigned lds_addr, const void* gptr,
    unsigned tensor_d0, unsigned tensor_d1,
    unsigned tile_d0, unsigned tile_d1, unsigned stride0) {
  unsigned long long ga = (unsigned long long)(uintptr_t)gptr;
  u32x4 g0;
  g0[0] = 1u;                                       // count=1 (valid), user mode
  g0[1] = lds_addr;                                 // lds_addr [63:32]
  g0[2] = (unsigned)(ga & 0xFFFFFFFFu);             // global_addr[31:0]
  g0[3] = (unsigned)((ga >> 32) & 0x01FFFFFFu)      // global_addr[56:32]
        | (2u << 30);                               // type=2 ("image")
  i32x8 g1;
  g1[0] = (int)(1u << 16);                          // data_size=1 -> 2 bytes
  g1[1] = (int)((tensor_d0 & 0xFFFFu) << 16);       // tensor_dim0[15:0] @63:48
  g1[2] = (int)(((tensor_d0 >> 16) & 0xFFFFu) |     // tensor_dim0[31:16]
                ((tensor_d1 & 0xFFFFu) << 16));     // tensor_dim1[15:0]
  g1[3] = (int)(((tensor_d1 >> 16) & 0xFFFFu) |     // tensor_dim1[31:16]
                ((tile_d0 & 0xFFFFu) << 16));       // tile_dim0 @127:112
  g1[4] = (int)(tile_d1 & 0xFFFFu);                 // tile_dim1; tile_dim2=0
  g1[5] = (int)stride0;                             // tensor_dim0_stride[31:0]
  g1[6] = 0;                                        // stride0[47:32], stride1 lo
  g1[7] = 0;
  i32x4 z4 = {0, 0, 0, 0};
#if defined(__clang_major__) && (__clang_major__ >= 23)
  i32x8 z8 = {0, 0, 0, 0, 0, 0, 0, 0};
  __builtin_amdgcn_tensor_load_to_lds(g0, g1, z4, z4, z8, 0);   // 6-arg form
#else
  __builtin_amdgcn_tensor_load_to_lds(g0, g1, z4, z4, 0);       // 5-arg form
#endif
}
#endif

// ---------------------------------------------------------------------------
// fp32 -> bf16 elementwise convert (grid-stride)
// ---------------------------------------------------------------------------
__global__ void f32_to_bf16_kernel(const float* __restrict__ in,
                                   bf16_t* __restrict__ out, int n) {
  int i = blockIdx.x * blockDim.x + threadIdx.x;
  int stride = gridDim.x * blockDim.x;
  for (; i < n; i += stride) out[i] = (bf16_t)in[i];
}

// ---------------------------------------------------------------------------
// bf16 GEMM: C[MxN] = A[MxK] * B[KxN], row-major, WMMA f32 accum.
// Block tile 128x128, BK=32, 256 threads = 8 waves (4M x 2N), each wave does
// 2x4 16x16 WMMA tiles per K-step. Double-buffered LDS; A tile moved by the
// Tensor Data Mover (wave 0 issues, s_wait_tensorcnt before barrier) so the
// DMA for tile k+1 overlaps WMMA on tile k. B staged transposed ([n][k]).
// ---------------------------------------------------------------------------
template <typename OutT>
__global__ __launch_bounds__(256) void gemm_bf16_wmma(
    const bf16_t* __restrict__ A, const bf16_t* __restrict__ B,
    OutT* __restrict__ C, int M, int N, int K) {
  __shared__ __align__(32) bf16_t sA[2][128][32];
  __shared__ __align__(32) bf16_t sB[2][128][32];  // transposed: [n][k]

  const int tid   = threadIdx.x;
  const int lane  = tid & 31;
  const int wave  = tid >> 5;
  const int waveM = (wave >> 1) * 32;  // 4 waves along M
  const int waveN = (wave & 1) * 64;   // 2 waves along N
  const int mBase = blockIdx.y * 128;
  const int nBase = blockIdx.x * 128;

  v8f acc[2][4] = {};

  auto loadA = [&](int buf, int k0) {
#if HAVE_TDM
    if (wave == 0) {
      tdm_load_2d_bf16((unsigned)(uintptr_t)&sA[buf][0][0],
                       A + (size_t)mBase * K + k0,
                       (unsigned)K, (unsigned)M, 32u, 128u, (unsigned)K);
    }
#else
    int row = tid >> 1;
    int col = (tid & 1) * 16;
    const bf16_t* src = A + (size_t)(mBase + row) * K + k0 + col;
    *(v16bf*)&sA[buf][row][col] = *(const v16bf*)src;
#endif
  };
  auto loadB = [&](int buf, int k0) {
    int kk = tid & 31;
    int n0 = (tid >> 5) * 16;
    const bf16_t* src = B + (size_t)(k0 + kk) * N + nBase + n0;
    v16bf vb = *(const v16bf*)src;
    if (k0 + 32 < K) __builtin_prefetch(src + (size_t)32 * N, 0, 1);
#pragma unroll
    for (int j = 0; j < 16; ++j) sB[buf][n0 + j][kk] = vb[j];
  };

  const int KT = K / 32;
  loadA(0, 0);
  loadB(0, 0);
#if HAVE_TDM
  if (wave == 0) __builtin_amdgcn_s_wait_tensorcnt(0);
#endif
  __syncthreads();

  for (int kt = 0; kt < KT; ++kt) {
    const int buf = kt & 1;
    if (kt + 1 < KT) {  // start DMA + global loads for next tile (overlaps WMMA)
      loadA(buf ^ 1, (kt + 1) * 32);
      loadB(buf ^ 1, (kt + 1) * 32);
    }

    const int ko = (lane >> 4) * 8;  // K-half per ISA lane layout
    Frag16 af[2], bfm[4];
#pragma unroll
    for (int i = 0; i < 2; ++i) {
      const bf16_t* p = &sA[buf][waveM + i * 16 + (lane & 15)][0];
      af[i].h[0] = *(const v8bf*)(p + ko);
      af[i].h[1] = *(const v8bf*)(p + 16 + ko);
    }
#pragma unroll
    for (int j = 0; j < 4; ++j) {
      const bf16_t* p = &sB[buf][waveN + j * 16 + (lane & 15)][0];
      bfm[j].h[0] = *(const v8bf*)(p + ko);
      bfm[j].h[1] = *(const v8bf*)(p + 16 + ko);
    }
#pragma unroll
    for (int i = 0; i < 2; ++i)
#pragma unroll
      for (int j = 0; j < 4; ++j)
        acc[i][j] = __builtin_amdgcn_wmma_f32_16x16x32_bf16(
            false, af[i].v, false, bfm[j].v, (short)0, acc[i][j], false, false);

#if HAVE_TDM
    if (wave == 0) __builtin_amdgcn_s_wait_tensorcnt(0);  // next A tile landed
#endif
    __syncthreads();
  }

  // C layout: lane holds rows (lane>>4)*8+r, col lane&15 of each 16x16 tile
#pragma unroll
  for (int i = 0; i < 2; ++i)
#pragma unroll
    for (int j = 0; j < 4; ++j)
#pragma unroll
      for (int r = 0; r < 8; ++r) {
        int row = mBase + waveM + i * 16 + (lane >> 4) * 8 + r;
        int col = nBase + waveN + j * 16 + (lane & 15);
        C[(size_t)row * N + col] = (OutT)acc[i][j][r];
      }
}

// ---------------------------------------------------------------------------
// RoPE in-place on bf16 [T, nh, dh]
// ---------------------------------------------------------------------------
__global__ void rope_kernel(bf16_t* __restrict__ x, int T, int nh, int dh) {
  int half  = dh >> 1;
  int total = T * nh * half;
  int idx = blockIdx.x * blockDim.x + threadIdx.x;
  if (idx >= total) return;
  int i = idx % half;
  int h = (idx / half) % nh;
  int t = idx / (half * nh);
  float inv = __powf(10000.0f, -(float)i / (float)half);
  float ang = (float)t * inv;
  float s, c;
  __sincosf(ang, &s, &c);
  bf16_t* p = x + ((size_t)t * nh + h) * dh;
  float x1 = (float)p[i], x2 = (float)p[i + half];
  p[i]        = (bf16_t)(x1 * c - x2 * s);
  p[i + half] = (bf16_t)(x1 * s + x2 * c);
}

// ---------------------------------------------------------------------------
// Flash attention (causal, GQA). Grid: (T/64, NH). 128 threads = 4 waves,
// each wave owns 16 query rows. K tile + V^T tile staged in LDS; P restaged
// through per-wave LDS to convert WMMA C-layout -> A-layout for PV.
// ---------------------------------------------------------------------------
__global__ __launch_bounds__(128) void flash_attn_wmma(
    const bf16_t* __restrict__ q,   // [T, NH*DH]
    const bf16_t* __restrict__ k,   // [T, NKV*DH]
    const bf16_t* __restrict__ v,   // [T, NKV*DH]
    bf16_t* __restrict__ y)         // [T, NH*DH]
{
  __shared__ __align__(32) bf16_t sK[64][DH_];      // 16 KB
  __shared__ __align__(32) bf16_t sVt[DH_][64];     // 16 KB (transposed)
  __shared__ __align__(32) bf16_t sP[4][16][64];    // 8 KB (per-wave P)

  const int lane  = threadIdx.x & 31;
  const int wave  = threadIdx.x >> 5;
  const int h     = blockIdx.y;
  const int kvh   = h / GRP_;
  const int qBlk  = blockIdx.x;
  const int qRow0 = qBlk * 64 + wave * 16;
  const int ko    = (lane >> 4) * 8;          // K-half offset for operands
  const float scale  = 0.08838834764831845f;  // 1/sqrt(128)
  const float NEGINF = -__builtin_inff();

  // Q fragments (A layout), 16 rows x 128 dh, resident in VGPRs
  Frag16 qf[4];
  {
    const bf16_t* base = q + (size_t)(qRow0 + (lane & 15)) * D_ + h * DH_;
#pragma unroll
    for (int kk = 0; kk < 4; ++kk) {
      qf[kk].h[0] = *(const v8bf*)(base + kk * 32 + ko);
      qf[kk].h[1] = *(const v8bf*)(base + kk * 32 + 16 + ko);
    }
  }

  float m[8], l[8];
  v8f o[8] = {};
#pragma unroll
  for (int r = 0; r < 8; ++r) { m[r] = NEGINF; l[r] = 0.0f; }

  const int nKB = qBlk + 1;  // causal: only key blocks <= query block
  for (int kb = 0; kb < nKB; ++kb) {
    const int kBase = kb * 64;
    // cooperative load K tile (row-major)
    {
      int row = threadIdx.x >> 1;
      int c0  = (threadIdx.x & 1) * 64;
      const bf16_t* src = k + (size_t)(kBase + row) * KVD_ + kvh * DH_ + c0;
#pragma unroll
      for (int j = 0; j < 4; ++j)
        *(v16bf*)&sK[row][c0 + j * 16] = *(const v16bf*)(src + j * 16);
    }
    // cooperative load V tile, transposed into sVt[dh][key]
    {
      int key = threadIdx.x & 63;
      int d0  = (threadIdx.x >> 6) * 64;
      const bf16_t* src = v + (size_t)(kBase + key) * KVD_ + kvh * DH_ + d0;
#pragma unroll
      for (int d = 0; d < 64; ++d) sVt[d0 + d][key] = src[d];
    }
    __syncthreads();

    // S = Q K^T : 4 key tiles x 4 K-steps of WMMA
    v8f s[4] = {};
#pragma unroll
    for (int nt = 0; nt < 4; ++nt) {
#pragma unroll
      for (int kk = 0; kk < 4; ++kk) {
        const bf16_t* p = &sK[nt * 16 + (lane & 15)][kk * 32];
        Frag16 bfrag;
        bfrag.h[0] = *(const v8bf*)(p + ko);
        bfrag.h[1] = *(const v8bf*)(p + 16 + ko);
        s[nt] = __builtin_amdgcn_wmma_f32_16x16x32_bf16(
            false, qf[kk].v, false, bfrag.v, (short)0, s[nt], false, false);
      }
    }

    // scale + causal mask (C layout: row = qRow0 + (lane>>4)*8+r, col = lane&15)
#pragma unroll
    for (int nt = 0; nt < 4; ++nt)
#pragma unroll
      for (int r = 0; r < 8; ++r) {
        int row = qRow0 + (lane >> 4) * 8 + r;
        int col = kBase + nt * 16 + (lane & 15);
        float val = s[nt][r] * scale;
        s[nt][r] = (col <= row) ? val : NEGINF;
      }

    // online softmax; row reductions across the 16 lanes of each half
#pragma unroll
    for (int r = 0; r < 8; ++r) {
      float mloc = NEGINF;
#pragma unroll
      for (int nt = 0; nt < 4; ++nt) mloc = fmaxf(mloc, s[nt][r]);
#pragma unroll
      for (int off = 8; off >= 1; off >>= 1)
        mloc = fmaxf(mloc, __shfl_xor(mloc, off, 32));
      float mnew  = fmaxf(m[r], mloc);
      float alpha = __expf(m[r] - mnew);  // exp(-inf)=0 on first block
      float sum = 0.0f;
      int prow = (lane >> 4) * 8 + r;
#pragma unroll
      for (int nt = 0; nt < 4; ++nt) {
        float p = __expf(s[nt][r] - mnew);
        sum += p;
        sP[wave][prow][nt * 16 + (lane & 15)] = (bf16_t)p;
      }
#pragma unroll
      for (int off = 8; off >= 1; off >>= 1) sum += __shfl_xor(sum, off, 32);
      l[r] = l[r] * alpha + sum;
      m[r] = mnew;
#pragma unroll
      for (int j = 0; j < 8; ++j) o[j][r] *= alpha;
    }
    // per-wave LDS region; same-wave DS ops are in-order -> no barrier needed

    // O += P V : 8 dh tiles x 2 K-steps (64 keys)
#pragma unroll
    for (int j = 0; j < 8; ++j) {
#pragma unroll
      for (int kk = 0; kk < 2; ++kk) {
        const bf16_t* pa = &sP[wave][lane & 15][kk * 32];
        const bf16_t* pb = &sVt[j * 16 + (lane & 15)][kk * 32];
        Frag16 afrag, bfrag;
        afrag.h[0] = *(const v8bf*)(pa + ko);
        afrag.h[1] = *(const v8bf*)(pa + 16 + ko);
        bfrag.h[0] = *(const v8bf*)(pb + ko);
        bfrag.h[1] = *(const v8bf*)(pb + 16 + ko);
        o[j] = __builtin_amdgcn_wmma_f32_16x16x32_bf16(
            false, afrag.v, false, bfrag.v, (short)0, o[j], false, false);
      }
    }
    __syncthreads();
  }

  // normalize and write y (bf16)
#pragma unroll
  for (int j = 0; j < 8; ++j)
#pragma unroll
    for (int r = 0; r < 8; ++r) {
      int row = qRow0 + (lane >> 4) * 8 + r;
      int dh0 = j * 16 + (lane & 15);
      y[(size_t)row * D_ + h * DH_ + dh0] = (bf16_t)(o[j][r] / l[r]);
    }
}

// ---------------------------------------------------------------------------
extern "C" void kernel_launch(void* const* d_in, const int* in_sizes, int n_in,
                              void* d_out, int out_size, void* d_ws, size_t ws_size,
                              hipStream_t stream) {
  const float* x  = (const float*)d_in[0];
  const float* Wq = (const float*)d_in[1];
  const float* Wk = (const float*)d_in[2];
  const float* Wv = (const float*)d_in[3];
  const float* Wo = (const float*)d_in[4];
  float* out = (float*)d_out;

  char* ws = (char*)d_ws;
  bf16_t* xb   = (bf16_t*)ws; ws += (size_t)T_ * D_ * 2;
  bf16_t* Wqb  = (bf16_t*)ws; ws += (size_t)D_ * D_ * 2;
  bf16_t* Wkb  = (bf16_t*)ws; ws += (size_t)D_ * KVD_ * 2;
  bf16_t* Wvb  = (bf16_t*)ws; ws += (size_t)D_ * KVD_ * 2;
  bf16_t* Wob  = (bf16_t*)ws; ws += (size_t)D_ * D_ * 2;
  bf16_t* qb   = (bf16_t*)ws; ws += (size_t)T_ * D_ * 2;
  bf16_t* kbuf = (bf16_t*)ws; ws += (size_t)T_ * KVD_ * 2;
  bf16_t* vbuf = (bf16_t*)ws; ws += (size_t)T_ * KVD_ * 2;
  bf16_t* yb   = (bf16_t*)ws; ws += (size_t)T_ * D_ * 2;   // ~76 MB total

  auto cvt = [&](const float* src, bf16_t* dst, int n) {
    int blocks = (n + 255) / 256;
    if (blocks > 4096) blocks = 4096;
    f32_to_bf16_kernel<<<blocks, 256, 0, stream>>>(src, dst, n);
  };
  cvt(x,  xb,  T_ * D_);
  cvt(Wq, Wqb, D_ * D_);
  cvt(Wk, Wkb, D_ * KVD_);
  cvt(Wv, Wvb, D_ * KVD_);
  cvt(Wo, Wob, D_ * D_);

  dim3 blk(256);
  gemm_bf16_wmma<bf16_t><<<dim3(D_ / 128,   T_ / 128), blk, 0, stream>>>(xb, Wqb, qb,   T_, D_,   D_);
  gemm_bf16_wmma<bf16_t><<<dim3(KVD_ / 128, T_ / 128), blk, 0, stream>>>(xb, Wkb, kbuf, T_, KVD_, D_);
  gemm_bf16_wmma<bf16_t><<<dim3(KVD_ / 128, T_ / 128), blk, 0, stream>>>(xb, Wvb, vbuf, T_, KVD_, D_);

  rope_kernel<<<(T_ * NH_  * (DH_ / 2) + 255) / 256, 256, 0, stream>>>(qb,   T_, NH_,  DH_);
  rope_kernel<<<(T_ * NKV_ * (DH_ / 2) + 255) / 256, 256, 0, stream>>>(kbuf, T_, NKV_, DH_);

  flash_attn_wmma<<<dim3(T_ / 64, NH_), dim3(128), 0, stream>>>(qb, kbuf, vbuf, yb);

  gemm_bf16_wmma<float><<<dim3(D_ / 128, T_ / 128), blk, 0, stream>>>(yb, Wob, out, T_, D_, D_);
}
